// NCAModel_5334349381907
// MI455X (gfx1250) — compile-verified, matrix operand order checked by reading, not run on hardware
//
#include <hip/hip_runtime.h>

// MI455X / gfx1250, wave32. WMMA f16 16x16x32, f32 accumulate.

typedef __attribute__((ext_vector_type(16))) _Float16 v16h;
typedef __attribute__((ext_vector_type(8)))  float    v8f;

#define B_N  8
#define C_N  16
#define H_N  256
#define W_N  256
#define HID  128
#define D_N  48     // 3*C perception features
#define KPAD 64     // D_N padded to 2 x 32 for WMMA K-slices

// CDNA5 WMMA 16-bit operand K mapping (ISA 7.12.2):
// lanes 0-15 hold K = {0..7, 16..23}; lanes 16-31 hold K = {8..15, 24..31}.
__device__ __forceinline__ int kmap(int i, int half) {
    return (i < 8 ? i : i + 8) + (half << 3);
}

// ---------------------------------------------------------------------------
// Kernel A: perception + MLP + stochastic residual update.
// One block = one (batch, row). 128 threads = 4 waves, each wave owns
// 4 groups of 16 pixels -> per group: 16 WMMA (fc0) + 4 WMMA (fc1).
// ---------------------------------------------------------------------------
__global__ __launch_bounds__(128)
void nca_update_kernel(const float* __restrict__ x,
                       const float* __restrict__ fc0_w,   // [128][48]
                       const float* __restrict__ fc0_b,   // [128]
                       const float* __restrict__ fc1_w,   // [16][128]
                       const float* __restrict__ stoch_t, // [B][C][H]
                       float* __restrict__ mid)           // x + dd*mask
{
    __shared__ _Float16 s_rows[3][C_N][W_N];   // 24 KB: rows r-1,r,r+1 (wrap)
    __shared__ _Float16 s_w0[HID][KPAD];       // 16 KB: fc0_w, K zero-padded
    __shared__ _Float16 s_w1[C_N][HID];        //  4 KB: fc1_w
    __shared__ float    s_bias[HID];           //  0.5 KB
    __shared__ _Float16 s_stage[4][16][HID];   // 16 KB: per-wave d (cols 0..63) then h (0..127)

    const int tid  = threadIdx.x;
    const int lane = tid & 31;
    const int wave = tid >> 5;
    const int r    = blockIdx.x;
    const int b    = blockIdx.y;
    const int p    = lane & 15;   // matrix row (pixel in group) for A fragments
    const int half = lane >> 4;   // lane half -> K interleave selector
    const int chN  = lane & 15;   // matrix column (channel / hidden unit) for B/C/D

    // --- stage LDS: 3 wrapped rows of all channels, plus weights -----------
    for (int idx = tid; idx < 3 * C_N * W_N; idx += 128) {
        int dr  = idx / (C_N * W_N);
        int rem = idx - dr * (C_N * W_N);
        int c   = rem >> 8;
        int w   = rem & (W_N - 1);
        int rr  = (r + dr + H_N - 1) & (H_N - 1);
        s_rows[dr][c][w] =
            (_Float16)x[(((size_t)b * C_N + c) * H_N + rr) * W_N + w];
    }
    for (int idx = tid; idx < HID * KPAD; idx += 128) {
        int n = idx >> 6, k = idx & 63;
        s_w0[n][k] = (_Float16)(k < D_N ? fc0_w[n * D_N + k] : 0.0f);
    }
    for (int idx = tid; idx < C_N * HID; idx += 128) {
        int n = idx >> 7, k = idx & 127;
        s_w1[n][k] = (_Float16)fc1_w[n * HID + k];
    }
    if (tid < HID) s_bias[tid] = fc0_b[tid];
    __syncthreads();

    // stochastic fire mask: depends on (b, channel, row) only, broadcast over W
    const float msk =
        (stoch_t[((size_t)b * C_N + chN) * H_N + r] > 0.5f) ? 1.0f : 0.0f;

    for (int g = 0; g < 4; ++g) {
        const int w0 = ((wave << 2) + g) << 4;   // first pixel of this 16-group
        const int w  = w0 + p;
        const int wm = (w + W_N - 1) & (W_N - 1);
        const int wp = (w + 1) & (W_N - 1);

        // --- perception: d = [x, sobel_x, sobel_y], K padded to 64 ---------
        #pragma unroll
        for (int cc = 0; cc < 8; ++cc) {
            const int ch = cc + (half << 3);
            float a00 = (float)s_rows[0][ch][wm], a01 = (float)s_rows[0][ch][w], a02 = (float)s_rows[0][ch][wp];
            float a10 = (float)s_rows[1][ch][wm], a11 = (float)s_rows[1][ch][w], a12 = (float)s_rows[1][ch][wp];
            float a20 = (float)s_rows[2][ch][wm], a21 = (float)s_rows[2][ch][w], a22 = (float)s_rows[2][ch][wp];
            float y1 = ((a02 - a00) + 2.0f * (a12 - a10) + (a22 - a20)) * 0.125f;
            float y2 = ((a20 + 2.0f * a21 + a22) - (a00 + 2.0f * a01 + a02)) * 0.125f;
            s_stage[wave][p][ch]      = (_Float16)a11;
            s_stage[wave][p][16 + ch] = (_Float16)y1;
            s_stage[wave][p][32 + ch] = (_Float16)y2;
            s_stage[wave][p][48 + ch] = (_Float16)0.0f;  // K pad
        }
        asm volatile("s_wait_dscnt 0" ::: "memory");  // intra-wave stage handoff

        // --- A fragments for fc0: 16 pixels x (2 x K32) --------------------
        v16h a0, a1;
        #pragma unroll
        for (int i = 0; i < 16; ++i) {
            a0[i] = s_stage[wave][p][kmap(i, half)];
            a1[i] = s_stage[wave][p][32 + kmap(i, half)];
        }

        // --- h = relu(d @ fc0_w^T + b): 8 N-tiles of 16 --------------------
        #pragma unroll
        for (int nt = 0; nt < 8; ++nt) {
            v8f c;
            const float bv = s_bias[(nt << 4) + chN];
            #pragma unroll
            for (int j = 0; j < 8; ++j) c[j] = bv;
            v16h b0, b1;
            #pragma unroll
            for (int i = 0; i < 16; ++i) {
                b0[i] = s_w0[(nt << 4) + chN][kmap(i, half)];
                b1[i] = s_w0[(nt << 4) + chN][32 + kmap(i, half)];
            }
            c = __builtin_amdgcn_wmma_f32_16x16x32_f16(false, a0, false, b0,
                                                       (short)0, c, false, false);
            c = __builtin_amdgcn_wmma_f32_16x16x32_f16(false, a1, false, b1,
                                                       (short)0, c, false, false);
            // D layout: lane<16 -> M=j, lane>=16 -> M=j+8; column = chN.
            #pragma unroll
            for (int j = 0; j < 8; ++j) {
                const int m = j + (half << 3);
                s_stage[wave][m][(nt << 4) + chN] = (_Float16)fmaxf(c[j], 0.0f);
            }
        }
        asm volatile("s_wait_dscnt 0" ::: "memory");

        // --- dd = h @ fc1_w^T: K = 128 -> 4 K-slices, single N-tile --------
        v8f acc;
        #pragma unroll
        for (int j = 0; j < 8; ++j) acc[j] = 0.0f;
        #pragma unroll
        for (int ks = 0; ks < 4; ++ks) {
            v16h ah, bh;
            #pragma unroll
            for (int i = 0; i < 16; ++i) {
                ah[i] = s_stage[wave][p][(ks << 5) + kmap(i, half)];
                bh[i] = s_w1[chN][(ks << 5) + kmap(i, half)];
            }
            acc = __builtin_amdgcn_wmma_f32_16x16x32_f16(false, ah, false, bh,
                                                         (short)0, acc, false, false);
        }

        // --- mid = x + dd * mask -------------------------------------------
        #pragma unroll
        for (int j = 0; j < 8; ++j) {
            const int m  = j + (half << 3);
            const int wq = w0 + m;
            const float xv = (float)s_rows[1][chN][wq];
            mid[(((size_t)b * C_N + chN) * H_N + r) * W_N + wq] = xv + acc[j] * msk;
        }
    }
}

// ---------------------------------------------------------------------------
// Kernel B: alive masking. life = (max3x3(src ch0) > .1) & (max3x3(mid ch0) > .1)
// dst = mid * life.  Memory-bound; whole state lives in L2.
// ---------------------------------------------------------------------------
__global__ __launch_bounds__(256)
void nca_life_kernel(const float* __restrict__ src,
                     const float* __restrict__ mid,
                     float* __restrict__ dst)
{
    const int w = threadIdx.x;
    const int r = blockIdx.x;
    const int b = blockIdx.y;
    const int rm = (r + H_N - 1) & (H_N - 1), rp = (r + 1) & (H_N - 1);
    const int wm = (w + W_N - 1) & (W_N - 1), wp = (w + 1) & (W_N - 1);
    const size_t base0 = (size_t)b * C_N * H_N * W_N;  // channel-0 plane

    const int rs[3] = {rm, r, rp};
    const int ws[3] = {wm, w, wp};
    float mpre = -1e30f, mpost = -1e30f;
    #pragma unroll
    for (int i = 0; i < 3; ++i) {
        #pragma unroll
        for (int j = 0; j < 3; ++j) {
            const size_t o = base0 + (size_t)rs[i] * W_N + ws[j];
            mpre  = fmaxf(mpre,  src[o]);
            mpost = fmaxf(mpost, mid[o]);
        }
    }
    const float life = (mpre > 0.1f && mpost > 0.1f) ? 1.0f : 0.0f;

    #pragma unroll
    for (int c = 0; c < C_N; ++c) {
        const size_t idx = (((size_t)b * C_N + c) * H_N + r) * W_N + w;
        dst[idx] = mid[idx] * life;
    }
}

// ---------------------------------------------------------------------------
// Host launcher. Output layout: [3][B][C][H][W]. Ping-pong through the output
// slices (out2 = step-0 temp, out0 = step-1 temp), snapshot input into out0
// last. steps is fixed at 2 (matches setup_inputs).
// ---------------------------------------------------------------------------
extern "C" void kernel_launch(void* const* d_in, const int* in_sizes, int n_in,
                              void* d_out, int out_size, void* d_ws, size_t ws_size,
                              hipStream_t stream) {
    (void)in_sizes; (void)n_in; (void)out_size; (void)d_ws; (void)ws_size;

    const float* x     = (const float*)d_in[0];
    const float* fc0_w = (const float*)d_in[1];
    const float* fc0_b = (const float*)d_in[2];
    const float* fc1_w = (const float*)d_in[3];
    const float* stoch = (const float*)d_in[4];
    float* out = (float*)d_out;

    const size_t S   = (size_t)B_N * C_N * H_N * W_N;  // one state tensor
    const size_t SST = (size_t)B_N * C_N * H_N;        // per-step stoch stride

    dim3 gridA(H_N, B_N), blockA(128);
    dim3 gridB(H_N, B_N), blockB(256);

    // step 0: src = input x, mid -> out[2] (scratch), x1 -> out[1]
    nca_update_kernel<<<gridA, blockA, 0, stream>>>(
        x, fc0_w, fc0_b, fc1_w, stoch, out + 2 * S);
    nca_life_kernel<<<gridB, blockB, 0, stream>>>(x, out + 2 * S, out + 1 * S);

    // step 1: src = x1, mid -> out[0] (scratch), x2 -> out[2]
    nca_update_kernel<<<gridA, blockA, 0, stream>>>(
        out + 1 * S, fc0_w, fc0_b, fc1_w, stoch + SST, out + 0 * S);
    nca_life_kernel<<<gridB, blockB, 0, stream>>>(
        out + 1 * S, out + 0 * S, out + 2 * S);

    // out[0] = original input snapshot
    hipMemcpyAsync(out, x, S * sizeof(float), hipMemcpyDeviceToDevice, stream);
}
